// GeneralizedBilinear_46488726012320
// MI455X (gfx1250) — compile-verified, matrix operand order checked by reading, not run on hardware
//
#include <hip/hip_runtime.h>

typedef __attribute__((ext_vector_type(16))) _Float16 v16h;
typedef __attribute__((ext_vector_type(8)))  _Float16 v8h;
typedef __attribute__((ext_vector_type(8)))  float    v8f;
typedef __attribute__((ext_vector_type(4)))  float    f32x4;

#define BT  16      // B_*T_ (S_=1)
#define HW  16384   // H_*W_
#define KD  128     // IN1 (contraction dim of the big GEMM)
#define JD  128     // IN2 (contraction dim of the code)
#define ND  512     // OUT
#define MT  128     // M tile per workgroup
#define NT  128     // N tile per workgroup
#define LDT 136     // LDS row stride in halves: 272B = 17*16B (aligned, bank-rotating)

// ---------------------------------------------------------------------------
// Stage 1: per-(bt) effective weight Meff[o][i] = sum_j x2[j]*W[o,i,j] + B[o,i]
// (cast to f16) and effective bias beff[o] = x2 . A[o] + bias[o] (kept f32).
// One block = one (bt, o); 128 threads = one i each.
// ---------------------------------------------------------------------------
__global__ void build_meff(const float* __restrict__ x2,    // [BT][JD]
                           const float* __restrict__ W,     // [ND][KD][JD]
                           const float* __restrict__ Bm,    // [ND][KD]
                           const float* __restrict__ A,     // [ND][JD]
                           const float* __restrict__ bias,  // [ND]
                           _Float16* __restrict__ meff,     // [BT][ND][KD]
                           float* __restrict__ beff)        // [BT][ND]
{
    const int o  = blockIdx.x;
    const int bt = blockIdx.y;
    const int i  = threadIdx.x;           // 0..127

    __shared__ float sx[JD];
    __shared__ float red[JD];
    sx[i] = x2[bt * JD + i];
    __syncthreads();

    const float* wrow = W + ((size_t)o * KD + i) * JD;
    float acc = 0.f;
#pragma unroll 8
    for (int j = 0; j < JD; j += 4) {
        f32x4 w4 = *(const f32x4*)(wrow + j);
        acc += sx[j] * w4.x + sx[j + 1] * w4.y + sx[j + 2] * w4.z + sx[j + 3] * w4.w;
    }
    acc += Bm[(size_t)o * KD + i];
    meff[((size_t)bt * ND + o) * KD + i] = (_Float16)acc;

    // bias code: tree-reduce x2[j]*A[o][j] across the block
    red[i] = sx[i] * A[(size_t)o * JD + i];
    __syncthreads();
    for (int s = 64; s > 0; s >>= 1) {
        if (i < s) red[i] += red[i + s];
        __syncthreads();
    }
    if (i == 0) beff[bt * ND + o] = red[0] + bias[o];
}

// ---------------------------------------------------------------------------
// Stage 2: out[bt][hw][o] = in1[bt][hw][:] (f32->f16) x Meff_bt[:][o] + beff[o]
// 16 GEMMs of 16384x128x512, f16 WMMA with f32 accumulate. Memory-bound.
// Block = 256 threads (8 waves), tile 128x128, full K=128 in LDS.
// ---------------------------------------------------------------------------
__global__ void __launch_bounds__(256)
bilinear_gemm(const float* __restrict__ in1,      // [BT][HW][KD] f32
              const _Float16* __restrict__ meff,  // [BT][ND][KD] f16 (row = n, col = k)
              const float* __restrict__ beff,     // [BT][ND]
              float* __restrict__ out)            // [BT][HW][ND] f32
{
    __shared__ _Float16 Atile[MT * LDT];  // [m][k]
    __shared__ _Float16 Btile[NT * LDT];  // [n][k]  (== Meff layout)

    const int m0 = blockIdx.x * MT;
    const int n0 = blockIdx.y * NT;
    const int bt = blockIdx.z;
    const int t  = threadIdx.x;

    // ---- cooperative loads: A (f32 -> f16) and B (f16 copy) ----
    {
        const int r0 = t >> 5;   // 0..7
        const int c4 = t & 31;   // group of 4 elements along k
        const float* asrc = in1 + ((size_t)bt * HW + m0) * KD;
        const _Float16* bsrc = meff + ((size_t)bt * ND + n0) * KD;
#pragma unroll
        for (int p = 0; p < 16; ++p) {
            const int r = p * 8 + r0;
            f32x4 v = *(const f32x4*)(asrc + (size_t)r * KD + c4 * 4);
            union { _Float16 h[4]; uint2 u; } cv;
            cv.h[0] = (_Float16)v.x; cv.h[1] = (_Float16)v.y;
            cv.h[2] = (_Float16)v.z; cv.h[3] = (_Float16)v.w;
            *(uint2*)&Atile[r * LDT + c4 * 4] = cv.u;
        }
#pragma unroll
        for (int p = 0; p < 16; ++p) {
            const int r = p * 8 + r0;
            *(uint2*)&Btile[r * LDT + c4 * 4] =
                *(const uint2*)(bsrc + (size_t)r * KD + c4 * 4);
        }
    }
    __syncthreads();

    const int lane = t & 31;
    const int wave = t >> 5;
    const int r16  = lane & 15;       // row (A) / col (B) within 16x16 tile
    const int hi   = lane >> 4;       // half-wave selector
    const int am   = (wave & 3) * 32; // wave's M offset in tile (2 subtiles)
    const int bn   = (wave >> 2) * 64;// wave's N offset in tile (4 subtiles)

    v8f acc[2][4] = {};               // f32 accumulators, 16x16 each

    union Frag { v16h f; v8h h[2]; };

    // K loop: 4 steps of K=32. Per-lane f16 A/B fragment layout (ISA 7.12.2):
    // lanes 0-15 hold K = {kb..kb+7, kb+16..kb+23}, lanes 16-31 the +8 halves.
#pragma unroll
    for (int ks = 0; ks < 4; ++ks) {
        const int kb = ks * 32 + hi * 8;
        Frag a[2], b[4];
#pragma unroll
        for (int mi = 0; mi < 2; ++mi) {
            const _Float16* p = &Atile[(am + mi * 16 + r16) * LDT + kb];
            a[mi].h[0] = *(const v8h*)p;
            a[mi].h[1] = *(const v8h*)(p + 16);
        }
#pragma unroll
        for (int ni = 0; ni < 4; ++ni) {
            const _Float16* p = &Btile[(bn + ni * 16 + r16) * LDT + kb];
            b[ni].h[0] = *(const v8h*)p;
            b[ni].h[1] = *(const v8h*)(p + 16);
        }
#pragma unroll
        for (int mi = 0; mi < 2; ++mi)
#pragma unroll
            for (int ni = 0; ni < 4; ++ni)
                acc[mi][ni] = __builtin_amdgcn_wmma_f32_16x16x32_f16(
                    false, a[mi].f, false, b[ni].f,
                    (short)0, acc[mi][ni], false, false);
    }

    // ---- epilogue: add per-(bt,n) effective bias, store f32 ----
    // C/D layout: lane -> N = r16, VGPR v -> M = v + 8*hi.
#pragma unroll
    for (int ni = 0; ni < 4; ++ni) {
        const int n  = n0 + bn + ni * 16 + r16;
        const float bv = beff[bt * ND + n];
#pragma unroll
        for (int mi = 0; mi < 2; ++mi) {
            const int mbase = m0 + am + mi * 16 + hi * 8;
#pragma unroll
            for (int v = 0; v < 8; ++v) {
                out[((size_t)bt * HW + (mbase + v)) * ND + n] = acc[mi][ni][v] + bv;
            }
        }
    }
}

// ---------------------------------------------------------------------------
extern "C" void kernel_launch(void* const* d_in, const int* in_sizes, int n_in,
                              void* d_out, int out_size, void* d_ws, size_t ws_size,
                              hipStream_t stream) {
    const float* in1  = (const float*)d_in[0];  // (4,4,128,128,1,128)
    const float* in2  = (const float*)d_in[1];  // (4,4,1,128)
    const float* W    = (const float*)d_in[2];  // (512,128,128)
    const float* A    = (const float*)d_in[3];  // (512,128)
    const float* Bm   = (const float*)d_in[4];  // (512,128)
    const float* bias = (const float*)d_in[5];  // (512,)
    float* out = (float*)d_out;

    _Float16* meff = (_Float16*)d_ws;                                   // 2 MiB
    float*    beff = (float*)((char*)d_ws +
                              (size_t)BT * ND * KD * sizeof(_Float16)); // 32 KiB

    dim3 g1(ND, BT);
    build_meff<<<g1, 128, 0, stream>>>(in2, W, Bm, A, bias, meff, beff);

    dim3 g2(HW / MT, ND / NT, BT);
    bilinear_gemm<<<g2, 256, 0, stream>>>(in1, meff, beff, out);
}
